// DecoderLayer_7344394076654
// MI455X (gfx1250) — compile-verified, matrix operand order checked by reading, not run on hardware
//
#include <hip/hip_runtime.h>

// ---------------------------------------------------------------------------
// Decoder layer for MI455X (gfx1250): bf16 WMMA GEMMs + flash attention,
// TDM tensor_load_to_lds for GEMM A-tiles, async global->LDS for V-chunks.
// ---------------------------------------------------------------------------

typedef __bf16 bf16_t;
typedef __attribute__((ext_vector_type(8)))  __bf16 v8bf;
typedef __attribute__((ext_vector_type(16))) __bf16 v16bf;
typedef __attribute__((ext_vector_type(8)))  float  v8f;
typedef __attribute__((ext_vector_type(4)))  unsigned int u32x4;
typedef __attribute__((ext_vector_type(8)))  int  i32x8;
typedef __attribute__((ext_vector_type(4)))  int  i32x4;

static constexpr int Bk  = 16;
static constexpr int Uk  = 512;
static constexpr int Hk  = 8;
static constexpr int LQk = 512;
static constexpr int LKk = 512;
static constexpr int HIDk = 2048;
static constexpr int Dk  = 64;
static constexpr float EPSk = 1e-3f;

#if defined(__has_builtin)
#if __has_builtin(__builtin_amdgcn_tensor_load_to_lds)
#define HAVE_TDM 1
#endif
#endif
#ifndef HAVE_TDM
#define HAVE_TDM 0
#endif

static __device__ __forceinline__ v16bf ld_frag(const bf16_t* p0, const bf16_t* p1) {
    v8bf lo = *(const v8bf*)p0;
    v8bf hi = *(const v8bf*)p1;
    return __builtin_shufflevector(lo, hi, 0,1,2,3,4,5,6,7,8,9,10,11,12,13,14,15);
}

// butterfly reduce within 16-lane halves via ds_swizzle (imm XOR pattern)
template <int M>
static __device__ __forceinline__ float swz_xor(float x) {
    int i = __builtin_amdgcn_ds_swizzle(__float_as_int(x), (M << 10) | 0x1F);
    return __int_as_float(i);
}
static __device__ __forceinline__ float half_max(float x) {
    x = fmaxf(x, swz_xor<8>(x));
    x = fmaxf(x, swz_xor<4>(x));
    x = fmaxf(x, swz_xor<2>(x));
    x = fmaxf(x, swz_xor<1>(x));
    return x;
}
static __device__ __forceinline__ float half_sum(float x) {
    x += swz_xor<8>(x);
    x += swz_xor<4>(x);
    x += swz_xor<2>(x);
    x += swz_xor<1>(x);
    return x;
}

// async global->LDS (GV form), ASYNCcnt-tracked
static __device__ __forceinline__ void async_ld_b128(unsigned lds_off, const void* g) {
    unsigned long long a = (unsigned long long)g;
    asm volatile("global_load_async_to_lds_b128 %0, %1, off"
                 :: "v"(lds_off), "v"(a) : "memory");
}
static __device__ __forceinline__ void wait_async0() {
#if defined(__has_builtin) && __has_builtin(__builtin_amdgcn_s_wait_asynccnt)
    __builtin_amdgcn_s_wait_asynccnt(0);
#else
    asm volatile("s_wait_asynccnt 0x0" ::: "memory");
#endif
}

#if HAVE_TDM
// 2D TDM tile load: tile (tile_d0 x tile_d1) elements (2B), row stride = stride0.
static __device__ __forceinline__ void tdm_load_2d(unsigned lds_addr, const void* gaddr,
                                                   unsigned tile_d0, unsigned tile_d1,
                                                   unsigned tensor_d0, unsigned tensor_d1,
                                                   unsigned long long stride0) {
    unsigned long long ga = (unsigned long long)gaddr;
    u32x4 g0;
    g0[0] = 1u;                                   // count=1, user descriptor
    g0[1] = lds_addr;                             // LDS byte address
    g0[2] = (unsigned)(ga & 0xFFFFFFFFu);         // global_addr[31:0]
    g0[3] = (unsigned)((ga >> 32) & 0x01FFFFFFu) | (2u << 30);  // addr msbs | type=2
    i32x8 g1;
    g1[0] = (int)(1u << 16);                      // workgroup_mask=0, data_size=1 (2B)
    g1[1] = (int)((tensor_d0 & 0xFFFFu) << 16);   // tensor_dim0[15:0]
    g1[2] = (int)(((tensor_d0 >> 16) & 0xFFFFu) | ((tensor_d1 & 0xFFFFu) << 16));
    g1[3] = (int)(((tensor_d1 >> 16) & 0xFFFFu) | ((tile_d0 & 0xFFFFu) << 16));
    g1[4] = (int)(tile_d1 & 0xFFFFu);             // tile_dim1, tile_dim2=0
    g1[5] = (int)(stride0 & 0xFFFFFFFFull);       // tensor_dim0_stride[31:0]
    g1[6] = (int)((stride0 >> 32) & 0xFFFFull);   // stride0[47:32], stride1=0
    g1[7] = 0;
    i32x4 z4 = {0, 0, 0, 0};
#if __clang_major__ >= 23
    i32x8 z8 = {0, 0, 0, 0, 0, 0, 0, 0};
    __builtin_amdgcn_tensor_load_to_lds(g0, g1, z4, z4, z8, 0);
#else
    __builtin_amdgcn_tensor_load_to_lds(g0, g1, z4, z4, 0);
#endif
}
#endif

// ---------------------------------------------------------------------------
// fp32 -> bf16 elementwise convert (weights / source), grid-stride.
// ---------------------------------------------------------------------------
__global__ void cvt_bf16_kernel(const float* __restrict__ in, bf16_t* __restrict__ out, size_t n) {
    size_t i = (size_t)blockIdx.x * blockDim.x + threadIdx.x;
    size_t stride = (size_t)gridDim.x * blockDim.x;
    for (; i < n; i += stride) out[i] = (__bf16)in[i];
}

// ---------------------------------------------------------------------------
// LayerNorm over the units axis (stride-L access), unbiased std, bf16 out.
// ---------------------------------------------------------------------------
__global__ void __launch_bounds__(256) layernorm_kernel(const float* __restrict__ x,
                                                        const float* __restrict__ g,
                                                        const float* __restrict__ bta,
                                                        bf16_t* __restrict__ y, int L) {
    __shared__ float red[256];
    const int col = blockIdx.x;           // 0 .. B*L-1
    const int bidx = col / L;
    const int l = col - bidx * L;
    const int t = threadIdx.x;
    const size_t base = (size_t)bidx * Uk * L + l;

    float x0 = x[base + (size_t)t * L];
    float x1 = x[base + (size_t)(t + 256) * L];

    red[t] = x0 + x1;
    __syncthreads();
    #pragma unroll
    for (int s = 128; s > 0; s >>= 1) {
        if (t < s) red[t] += red[t + s];
        __syncthreads();
    }
    const float mean = red[0] * (1.0f / Uk);
    __syncthreads();

    float d0 = x0 - mean, d1 = x1 - mean;
    red[t] = d0 * d0 + d1 * d1;
    __syncthreads();
    #pragma unroll
    for (int s = 128; s > 0; s >>= 1) {
        if (t < s) red[t] += red[t + s];
        __syncthreads();
    }
    const float stdv = sqrtf(red[0] / (float)(Uk - 1));
    const float inv = 1.0f / (stdv + EPSk);

    y[base + (size_t)t * L]         = (__bf16)(g[t] * d0 * inv + bta[t]);
    y[base + (size_t)(t + 256) * L] = (__bf16)(g[t + 256] * d1 * inv + bta[t + 256]);
}

// ---------------------------------------------------------------------------
// Tiled bf16 WMMA GEMM:  out[b] = op( A(MxK) * X[b](KxN) )
//   MODE 0: -> bf16 | MODE 1: +bias, ReLU -> bf16 | MODE 2: +bias? +res -> f32
// Block: 256 threads = 8 waves; block tile 128x128; wave tile 64x32 (4x2 wmma).
// A tile streamed by the Tensor Data Mover when available.
// ---------------------------------------------------------------------------
template <int MODE>
__global__ void __launch_bounds__(256, 1) gemm_wmma_kernel(const bf16_t* __restrict__ A,
                                                           const bf16_t* __restrict__ X,
                                                           const float* bias,
                                                           const float* res,
                                                           float* outF,
                                                           bf16_t* outB,
                                                           int M, int K, int N) {
    __shared__ __align__(16) bf16_t As[128 * 32];
    __shared__ __align__(16) bf16_t Bs[128 * 32];

    const int tid  = threadIdx.x;
    const int lane = tid & 31;
    const int wid  = tid >> 5;
    const int wrow = wid >> 2;        // 0..1  (64 rows each)
    const int wcol = wid & 3;         // 0..3  (32 cols each)
    const int half = lane >> 4;       // K-striping half
    const int mr   = lane & 15;       // row (A) / col (B) within fragment

    const int m0 = blockIdx.y * 128;
    const int n0 = blockIdx.x * 128;
    const int batch = blockIdx.z;
    const bf16_t* Xb = X + (size_t)batch * K * N;
#if HAVE_TDM
    const unsigned as_lds = (unsigned)(size_t)(void*)As;
#endif

    const v8f vzero = {0.f, 0.f, 0.f, 0.f, 0.f, 0.f, 0.f, 0.f};
    v8f c[4][2];
    #pragma unroll
    for (int i = 0; i < 4; ++i)
        #pragma unroll
        for (int j = 0; j < 2; ++j) c[i][j] = vzero;

    for (int k0 = 0; k0 < K; k0 += 32) {
        // ---- stage A tile (128 x 32): TDM if available, manual otherwise ----
#if HAVE_TDM
        if (wid == 0) {
            tdm_load_2d(as_lds, A + (size_t)m0 * K + k0,
                        /*tile*/ 32u, 128u, /*tensor*/ (unsigned)K, (unsigned)M,
                        (unsigned long long)K);
        }
#else
        #pragma unroll
        for (int p = 0; p < 2; ++p) {
            int idx = p * 256 + tid;
            int row = idx >> 2;
            int kk  = (idx & 3) << 3;
            *(u32x4*)&As[row * 32 + kk] =
                *(const u32x4*)(A + (size_t)(m0 + row) * K + k0 + kk);
        }
#endif
        // ---- stage B tile transposed: Bs[n][k] = X[k0+k][n0+n] ----
        #pragma unroll
        for (int p = 0; p < 2; ++p) {
            int idx = p * 256 + tid;
            int kr = idx >> 4;
            int nn = (idx & 15) << 3;
            v8bf v = *(const v8bf*)(Xb + (size_t)(k0 + kr) * N + n0 + nn);
            #pragma unroll
            for (int j = 0; j < 8; ++j) Bs[(nn + j) * 32 + kr] = v[j];
        }
        if (k0 + 32 < K) {
            __builtin_prefetch((const void*)(Xb + (size_t)(k0 + 32 + (tid >> 4)) * N + n0), 0, 3);
        }
#if HAVE_TDM
        __builtin_amdgcn_s_wait_tensorcnt(0);
#endif
        __syncthreads();

        v16bf af[4], bfr[2];
        #pragma unroll
        for (int i = 0; i < 4; ++i) {
            int base = (wrow * 64 + i * 16 + mr) * 32 + half * 8;
            af[i] = ld_frag(&As[base], &As[base + 16]);
        }
        #pragma unroll
        for (int j = 0; j < 2; ++j) {
            int base = (wcol * 32 + j * 16 + mr) * 32 + half * 8;
            bfr[j] = ld_frag(&Bs[base], &Bs[base + 16]);
        }
        #pragma unroll
        for (int i = 0; i < 4; ++i)
            #pragma unroll
            for (int j = 0; j < 2; ++j)
                c[i][j] = __builtin_amdgcn_wmma_f32_16x16x32_bf16(
                    false, af[i], false, bfr[j], (short)0, c[i][j], false, false);
        __syncthreads();
    }

    // ---- epilogue ----
    #pragma unroll
    for (int i = 0; i < 4; ++i) {
        #pragma unroll
        for (int j = 0; j < 2; ++j) {
            #pragma unroll
            for (int v = 0; v < 8; ++v) {
                int row = m0 + wrow * 64 + i * 16 + half * 8 + v;
                int colg = n0 + wcol * 32 + j * 16 + mr;
                size_t oidx = (size_t)batch * M * N + (size_t)row * N + colg;
                float val = c[i][j][v];
                if constexpr (MODE == 1) {
                    val += bias[row];
                    val = fmaxf(val, 0.0f);
                    outB[oidx] = (__bf16)val;
                } else if constexpr (MODE == 2) {
                    if (bias) val += bias[row];
                    val += res[oidx];
                    outF[oidx] = val;
                } else {
                    outB[oidx] = (__bf16)val;
                }
            }
        }
    }
}

// ---------------------------------------------------------------------------
// Flash attention per (b, h). Block = 128 threads = 4 waves; each wave owns a
// 16-query tile. K chunk transposed in LDS (shared); V chunk streamed to LDS
// with async global->LDS loads (shared by all 4 waves). scale = 1/sqrt(D).
// ---------------------------------------------------------------------------
__global__ void __launch_bounds__(128, 1) attn_kernel(const bf16_t* __restrict__ Qb,
                                                      const bf16_t* __restrict__ Kb,
                                                      const bf16_t* __restrict__ Vb,
                                                      bf16_t* __restrict__ Cb,
                                                      int Lq, int Lk) {
    __shared__ __align__(16) bf16_t Qs[4][16 * 64];   // per-wave Q tile (q, d)
    __shared__ __align__(16) bf16_t Ks[32 * 64];      // shared K chunk (k, d)
    __shared__ __align__(16) bf16_t Vs[64 * 32];      // shared V chunk (d, k)
    __shared__ __align__(16) bf16_t Ps[4][16 * 32];   // per-wave P tile (q, k)

    const int tid  = threadIdx.x;
    const int lane = tid & 31;
    const int wid  = tid >> 5;
    const int half = lane >> 4;
    const int mr   = lane & 15;

    const int b = blockIdx.z, h = blockIdx.y;
    const int q0 = blockIdx.x * 64 + wid * 16;

    const bf16_t* Qh = Qb + ((size_t)b * Uk + h * Dk) * Lq;
    const bf16_t* Kh = Kb + ((size_t)b * Uk + h * Dk) * Lk;
    const bf16_t* Vh = Vb + ((size_t)b * Uk + h * Dk) * Lk;
    bf16_t*       Ch = Cb + ((size_t)b * Uk + h * Dk) * Lq;
    const unsigned vs_lds = (unsigned)(size_t)(void*)Vs;

    // stage Q tile transposed: Qs[q][d]   (wave-local)
    for (int idx = lane; idx < 16 * 64; idx += 32) {
        int d = idx >> 4, qi = idx & 15;
        Qs[wid][qi * 64 + d] = Qh[(size_t)d * Lq + q0 + qi];
    }

    v16bf aq[2];
    #pragma unroll
    for (int cch = 0; cch < 2; ++cch) {
        int base = mr * 64 + cch * 32 + half * 8;
        aq[cch] = ld_frag(&Qs[wid][base], &Qs[wid][base + 16]);
    }

    const v8f vzero = {0.f, 0.f, 0.f, 0.f, 0.f, 0.f, 0.f, 0.f};
    v8f o[4];
    #pragma unroll
    for (int t = 0; t < 4; ++t) o[t] = vzero;
    float mrun[8], srun[8];
    #pragma unroll
    for (int v = 0; v < 8; ++v) { mrun[v] = -3.0e38f; srun[v] = 0.0f; }

    const float scale = 0.125f;  // 1/sqrt(64)

    for (int k0 = 0; k0 < Lk; k0 += 32) {
        // stage K chunk transposed: Ks[k][d], cooperative across 128 threads
        for (int idx = tid; idx < 32 * 64; idx += 128) {
            int d = idx >> 5, kc = idx & 31;
            Ks[kc * 64 + d] = Kh[(size_t)d * Lk + k0 + kc];
        }
        // stage V chunk as-is via async global->LDS: Vs[d][k], 64x32 bf16
        #pragma unroll
        for (int p = 0; p < 2; ++p) {
            int idx = p * 128 + tid;          // 0..255 chunks of 16B
            int d  = idx >> 2;
            int kk = (idx & 3) << 3;
            async_ld_b128(vs_lds + (unsigned)(d * 32 + kk) * 2,
                          Vh + (size_t)d * Lk + k0 + kk);
        }
        wait_async0();
        __syncthreads();

        // S = (Q^T K) * scale : two 16x16 tiles (k cols 0-15, 16-31)
        v8f s[2];
        #pragma unroll
        for (int jk = 0; jk < 2; ++jk) {
            v8f acc = vzero;
            #pragma unroll
            for (int cch = 0; cch < 2; ++cch) {
                int base = (jk * 16 + mr) * 64 + cch * 32 + half * 8;
                v16bf bk = ld_frag(&Ks[base], &Ks[base + 16]);
                acc = __builtin_amdgcn_wmma_f32_16x16x32_bf16(
                    false, aq[cch], false, bk, (short)0, acc, false, false);
            }
            #pragma unroll
            for (int v = 0; v < 8; ++v) acc[v] *= scale;
            s[jk] = acc;
        }

        // online softmax per accumulator row (row = v + half*8 of this q tile)
        float corr[8];
        #pragma unroll
        for (int v = 0; v < 8; ++v) {
            float mnew = fmaxf(mrun[v], half_max(fmaxf(s[0][v], s[1][v])));
            float f = __expf(mrun[v] - mnew);
            float p0 = __expf(s[0][v] - mnew);
            float p1 = __expf(s[1][v] - mnew);
            float ls = half_sum(p0 + p1);
            srun[v] = srun[v] * f + ls;
            mrun[v] = mnew;
            corr[v] = f;
            int prow = v + half * 8;
            Ps[wid][prow * 32 + mr]      = (__bf16)p0;
            Ps[wid][prow * 32 + mr + 16] = (__bf16)p1;
        }
        #pragma unroll
        for (int t = 0; t < 4; ++t)
            #pragma unroll
            for (int v = 0; v < 8; ++v) o[t][v] *= corr[v];

        // O += P * V^T : P as A-matrix (16q x 32k), V columns from shared LDS
        {
            int base = mr * 32 + half * 8;
            v16bf ap = ld_frag(&Ps[wid][base], &Ps[wid][base + 16]);
            #pragma unroll
            for (int t = 0; t < 4; ++t) {
                int vb0 = (t * 16 + mr) * 32 + half * 8;
                v16bf bv = ld_frag(&Vs[vb0], &Vs[vb0 + 16]);
                o[t] = __builtin_amdgcn_wmma_f32_16x16x32_bf16(
                    false, ap, false, bv, (short)0, o[t], false, false);
            }
        }
        __syncthreads();
    }

    // normalize and store C (rows = d, cols = q)
    float rinv[8];
    #pragma unroll
    for (int v = 0; v < 8; ++v) rinv[v] = 1.0f / srun[v];
    #pragma unroll
    for (int t = 0; t < 4; ++t) {
        #pragma unroll
        for (int v = 0; v < 8; ++v) {
            int d = t * 16 + mr;
            int q = q0 + v + half * 8;
            Ch[(size_t)d * Lq + q] = (__bf16)(o[t][v] * rinv[v]);
        }
    }
}

// ---------------------------------------------------------------------------
// host-side orchestration
// ---------------------------------------------------------------------------
extern "C" void kernel_launch(void* const* d_in, const int* in_sizes, int n_in,
                              void* d_out, int out_size, void* d_ws, size_t ws_size,
                              hipStream_t stream) {
    const float* e     = (const float*)d_in[0];
    const float* src   = (const float*)d_in[1];
    const float* ln1g  = (const float*)d_in[2];
    const float* ln1b  = (const float*)d_in[3];
    const float* Wq1   = (const float*)d_in[4];
    const float* Wk1   = (const float*)d_in[5];
    const float* Wv1   = (const float*)d_in[6];
    const float* Wo1   = (const float*)d_in[7];
    const float* ln2g  = (const float*)d_in[8];
    const float* ln2b  = (const float*)d_in[9];
    const float* Wq2   = (const float*)d_in[10];
    const float* Wk2   = (const float*)d_in[11];
    const float* Wv2   = (const float*)d_in[12];
    const float* Wo2   = (const float*)d_in[13];
    const float* ln3g  = (const float*)d_in[14];
    const float* ln3b  = (const float*)d_in[15];
    const float* W1    = (const float*)d_in[16];
    const float* b1    = (const float*)d_in[17];
    const float* W2    = (const float*)d_in[18];
    const float* b2    = (const float*)d_in[19];
    (void)in_sizes; (void)n_in; (void)out_size; (void)ws_size;

    char* wsp = (char*)d_ws;
    auto alloc = [&](size_t bytes) -> void* {
        void* p = (void*)wsp;
        wsp += (bytes + 255) & ~(size_t)255;
        return p;
    };

    const size_t WW = (size_t)Uk * Uk;
    bf16_t* wq1b = (bf16_t*)alloc(WW * 2);
    bf16_t* wk1b = (bf16_t*)alloc(WW * 2);
    bf16_t* wv1b = (bf16_t*)alloc(WW * 2);
    bf16_t* wo1b = (bf16_t*)alloc(WW * 2);
    bf16_t* wq2b = (bf16_t*)alloc(WW * 2);
    bf16_t* wk2b = (bf16_t*)alloc(WW * 2);
    bf16_t* wv2b = (bf16_t*)alloc(WW * 2);
    bf16_t* wo2b = (bf16_t*)alloc(WW * 2);
    bf16_t* w1b  = (bf16_t*)alloc((size_t)HIDk * Uk * 2);
    bf16_t* w2b  = (bf16_t*)alloc((size_t)Uk * HIDk * 2);
    bf16_t* srcb = (bf16_t*)alloc((size_t)Bk * Uk * LKk * 2);
    bf16_t* xb   = (bf16_t*)alloc((size_t)Bk * Uk * LQk * 2);
    bf16_t* qb   = (bf16_t*)alloc((size_t)Bk * Uk * LQk * 2);
    bf16_t* kb   = (bf16_t*)alloc((size_t)Bk * Uk * LKk * 2);
    bf16_t* vb   = (bf16_t*)alloc((size_t)Bk * Uk * LKk * 2);
    bf16_t* cb   = (bf16_t*)alloc((size_t)Bk * Uk * LQk * 2);
    float*  E    = (float*)alloc((size_t)Bk * Uk * LQk * 4);
    bf16_t* hb   = (bf16_t*)alloc((size_t)Bk * HIDk * LQk * 4);

    auto cvt = [&](const float* s, bf16_t* dPtr, size_t n) {
        int blocks = (int)((n + 255) / 256);
        if (blocks > 4096) blocks = 4096;
        cvt_bf16_kernel<<<blocks, 256, 0, stream>>>(s, dPtr, n);
    };
    cvt(Wq1, wq1b, WW); cvt(Wk1, wk1b, WW); cvt(Wv1, wv1b, WW); cvt(Wo1, wo1b, WW);
    cvt(Wq2, wq2b, WW); cvt(Wk2, wk2b, WW); cvt(Wv2, wv2b, WW); cvt(Wo2, wo2b, WW);
    cvt(W1, w1b, (size_t)HIDk * Uk);
    cvt(W2, w2b, (size_t)Uk * HIDk);
    cvt(src, srcb, (size_t)Bk * Uk * LKk);

    const dim3 gProj(LQk / 128, Uk / 128, Bk);      // 4 x 4 x 16
    const dim3 gFf1(LQk / 128, HIDk / 128, Bk);     // 4 x 16 x 16
    const dim3 gAttn(LQk / 64, Hk, Bk);             // 8 x 8 x 16

    // ---- self-attention sublayer ----
    layernorm_kernel<<<Bk * LQk, 256, 0, stream>>>(e, ln1g, ln1b, xb, LQk);
    gemm_wmma_kernel<0><<<gProj, 256, 0, stream>>>(wq1b, xb, nullptr, nullptr, nullptr, qb, Uk, Uk, LQk);
    gemm_wmma_kernel<0><<<gProj, 256, 0, stream>>>(wk1b, xb, nullptr, nullptr, nullptr, kb, Uk, Uk, LQk);
    gemm_wmma_kernel<0><<<gProj, 256, 0, stream>>>(wv1b, xb, nullptr, nullptr, nullptr, vb, Uk, Uk, LQk);
    attn_kernel<<<gAttn, 128, 0, stream>>>(qb, kb, vb, cb, LQk, LQk);
    gemm_wmma_kernel<2><<<gProj, 256, 0, stream>>>(wo1b, cb, nullptr, e, E, nullptr, Uk, Uk, LQk);

    // ---- cross-attention sublayer ----
    layernorm_kernel<<<Bk * LQk, 256, 0, stream>>>(E, ln2g, ln2b, xb, LQk);
    gemm_wmma_kernel<0><<<gProj, 256, 0, stream>>>(wq2b, xb,   nullptr, nullptr, nullptr, qb, Uk, Uk, LQk);
    gemm_wmma_kernel<0><<<gProj, 256, 0, stream>>>(wk2b, srcb, nullptr, nullptr, nullptr, kb, Uk, Uk, LKk);
    gemm_wmma_kernel<0><<<gProj, 256, 0, stream>>>(wv2b, srcb, nullptr, nullptr, nullptr, vb, Uk, Uk, LKk);
    attn_kernel<<<gAttn, 128, 0, stream>>>(qb, kb, vb, cb, LQk, LKk);
    gemm_wmma_kernel<2><<<gProj, 256, 0, stream>>>(wo2b, cb, nullptr, E, E, nullptr, Uk, Uk, LQk);

    // ---- feed-forward sublayer ----
    layernorm_kernel<<<Bk * LQk, 256, 0, stream>>>(E, ln3g, ln3b, xb, LQk);
    gemm_wmma_kernel<1><<<gFf1, 256, 0, stream>>>(w1b, xb, b1, nullptr, nullptr, hb, HIDk, Uk, LQk);
    gemm_wmma_kernel<2><<<gProj, 256, 0, stream>>>(w2b, hb, b2, E, (float*)d_out, nullptr, Uk, HIDk, LQk);
}
